// NSVQ_30107720745643
// MI455X (gfx1250) — matrix operand order repaired and checked by reading, not compiled, not run on hardware
//
#include <hip/hip_runtime.h>
#include <hip/hip_bf16.h>

// ---------------------------------------------------------------------------
// Types for CDNA5 WMMA (wave32): 16x16x32 bf16 -> f32 accumulate
// ---------------------------------------------------------------------------
typedef __attribute__((ext_vector_type(16))) __bf16 bf16x16;
typedef __attribute__((ext_vector_type(8)))  __bf16 bf16x8;
typedef __attribute__((ext_vector_type(8)))  float  v8f;
typedef int v4i_vs __attribute__((vector_size(16)));   // matches async-LDS builtin pointee

__device__ __forceinline__ bf16x16 frag_join(bf16x8 lo, bf16x8 hi) {
    bf16x16 r;
#pragma unroll
    for (int i = 0; i < 8; ++i) { r[i] = lo[i]; r[i + 8] = hi[i]; }
    return r;
}

// A fragment (16x32, row-major A[M x K], lda = K).
__device__ __forceinline__ bf16x16 load_a_frag(const __bf16* A, int lda,
                                               int m0, int k0, int lane) {
    int row = m0 + (lane & 15);
    int kb  = k0 + ((lane >> 4) << 3);
    const __bf16* p = A + (size_t)row * lda + kb;
    bf16x8 lo = *reinterpret_cast<const bf16x8*>(p);
    bf16x8 hi = *reinterpret_cast<const bf16x8*>(p + 16);
    return frag_join(lo, hi);
}

// B fragment (32x16) from a TRANSPOSED B: Bt[N x K] row-major, ldb = K.
__device__ __forceinline__ bf16x16 load_bt_frag(const __bf16* Bt, int ldb,
                                                int n0, int k0, int lane) {
    int row = n0 + (lane & 15);
    int ks  = k0 + ((lane >> 4) << 3);
    const __bf16* p = Bt + (size_t)row * ldb + ks;
    bf16x8 lo = *reinterpret_cast<const bf16x8*>(p);
    bf16x8 hi = *reinterpret_cast<const bf16x8*>(p + 16);
    return frag_join(lo, hi);
}

__device__ __forceinline__ v8f wmma_bf16(bf16x16 a, bf16x16 b, v8f c) {
    return __builtin_amdgcn_wmma_f32_16x16x32_bf16(
        false, a, false, b, (short)0, c, false, false);
}

// ---------------------------------------------------------------------------
// Problem constants
// ---------------------------------------------------------------------------
#define NB   64
#define NT   256
#define NDIM 1024
#define NED  512
#define NNE  8192
#define NCODES 64
#define NROWS (NB * NCODES)   // 4096

// ---------------------------------------------------------------------------
// Elementwise conversion / prep kernels
// ---------------------------------------------------------------------------
__global__ void k_diff_cvt(const float* __restrict__ last,
                           const float* __restrict__ first,
                           __bf16* __restrict__ out, int n) {
    int i = blockIdx.x * blockDim.x + threadIdx.x;
    if (i < n) out[i] = (__bf16)(last[i] - first[i]);
}

// W_in [1024,512] -> WinT bf16 [512,1024]
__global__ void k_cvt_winT(const float* __restrict__ W, __bf16* __restrict__ Wt) {
    int i = blockIdx.x * blockDim.x + threadIdx.x;
    int k = i >> 9, n = i & 511;
    Wt[(size_t)n * NDIM + k] = (__bf16)W[i];
}

// conv_w [OC=512, IC=512, 3, 3] -> WcT bf16 [OC, 4608], kk = tap*512 + ic
__global__ void k_cvt_wcT(const float* __restrict__ W, __bf16* __restrict__ Wt) {
    int i = blockIdx.x * blockDim.x + threadIdx.x;
    int oc = i / 4608, r = i % 4608;
    int ic = r / 9, tap = r % 9;
    Wt[(size_t)oc * 4608 + tap * NED + ic] = (__bf16)W[i];
}

// W_out [512,1024] -> WoutT bf16 [1024,512]
__global__ void k_cvt_woutT(const float* __restrict__ W, __bf16* __restrict__ Wt) {
    int i = blockIdx.x * blockDim.x + threadIdx.x;
    int e = i >> 10, d = i & 1023;
    Wt[(size_t)d * NED + e] = (__bf16)W[i];
}

// codebooks [8192,512] -> bf16 + per-row squared norm
__global__ void k_cvt_cb(const float* __restrict__ C, __bf16* __restrict__ Ch,
                         float* __restrict__ cnorm) {
    int r = blockIdx.x * blockDim.x + threadIdx.x;
    if (r >= NNE) return;
    const float* src = C + (size_t)r * NED;
    __bf16* dst = Ch + (size_t)r * NED;
    float s = 0.f;
#pragma unroll 4
    for (int e = 0; e < NED; ++e) { float v = src[e]; s += v * v; dst[e] = (__bf16)v; }
    cnorm[r] = s;
}

__global__ void k_zero_hist(int* __restrict__ hist) {
    int i = blockIdx.x * blockDim.x + threadIdx.x;
    if (i < NNE) hist[i] = 0;
}

// ---------------------------------------------------------------------------
// Tile store helper
// ---------------------------------------------------------------------------
__device__ __forceinline__ void store_tile(float* Cf, __bf16* Ch, const float* bias,
                                           int N, int m0, int n0, int lane, v8f acc) {
    int col = n0 + (lane & 15);
    int rb  = m0 + ((lane >> 4) << 3);
    float bv = bias ? bias[col] : 0.f;
#pragma unroll
    for (int v = 0; v < 8; ++v) {
        float r = acc[v] + bv;
        if (Cf) Cf[(size_t)(rb + v) * N + col] = r;
        if (Ch) Ch[(size_t)(rb + v) * N + col] = (__bf16)r;
    }
}

// ---------------------------------------------------------------------------
// Generic WMMA GEMM, register-blocked: each wave -> 16(M) x 64(N) tile.
// A fragment is loaded once per k-step and reused across 4 WMMAs.
// ---------------------------------------------------------------------------
__global__ void k_gemm_bf16(const __bf16* __restrict__ A,
                            const __bf16* __restrict__ Bt,
                            float* __restrict__ Cf, __bf16* __restrict__ Ch,
                            const float* __restrict__ bias,
                            int M, int N, int K) {
    int wave = blockIdx.x * (blockDim.x >> 5) + (threadIdx.x >> 5);
    int lane = threadIdx.x & 31;
    int mt = M >> 4, ng = N >> 6;
    if (wave >= mt * ng) return;             // wave-uniform exit
    int m0 = (wave % mt) << 4;
    int n0 = (wave / mt) << 6;

    v8f acc0 = {}, acc1 = {}, acc2 = {}, acc3 = {};
    for (int k0 = 0; k0 < K; k0 += 32) {
        bf16x16 a = load_a_frag(A, K, m0, k0, lane);
        if (k0 + 64 < K) {
            // prefetch the A row two k-steps ahead (-> global_prefetch_b8)
            __builtin_prefetch(A + (size_t)(m0 + (lane & 15)) * K + k0 + 64, 0, 3);
        }
        bf16x16 b0 = load_bt_frag(Bt, K, n0,      k0, lane);
        bf16x16 b1 = load_bt_frag(Bt, K, n0 + 16, k0, lane);
        bf16x16 b2 = load_bt_frag(Bt, K, n0 + 32, k0, lane);
        bf16x16 b3 = load_bt_frag(Bt, K, n0 + 48, k0, lane);
        acc0 = wmma_bf16(a, b0, acc0);
        acc1 = wmma_bf16(a, b1, acc1);
        acc2 = wmma_bf16(a, b2, acc2);
        acc3 = wmma_bf16(a, b3, acc3);
    }
    store_tile(Cf, Ch, bias, N, m0, n0,      lane, acc0);
    store_tile(Cf, Ch, bias, N, m0, n0 + 16, lane, acc1);
    store_tile(Cf, Ch, bias, N, m0, n0 + 32, lane, acc2);
    store_tile(Cf, Ch, bias, N, m0, n0 + 48, lane, acc3);
}

// ---------------------------------------------------------------------------
// 3x3 stride-2 pad-1 conv on [B,512,16,16] as implicit GEMM (M=4096,K=4608,N=512).
// The im2col A gather is done once per k-step and reused across 4 N-tiles.
// ---------------------------------------------------------------------------
__global__ void k_conv_gemm(const __bf16* __restrict__ Y,
                            const __bf16* __restrict__ WcT,
                            float* __restrict__ X, __bf16* __restrict__ Xh) {
    const int MT = NROWS >> 4;               // 256
    const int NG = NED >> 6;                 // 8
    int wave = blockIdx.x * (blockDim.x >> 5) + (threadIdx.x >> 5);
    int lane = threadIdx.x & 31;
    if (wave >= MT * NG) return;
    int m0 = (wave % MT) << 4;
    int n0 = (wave / MT) << 6;

    int rowm = m0 + (lane & 15);
    int b = rowm >> 6, pix = rowm & 63;
    int oh = pix >> 3, ow = pix & 7;

    v8f acc0 = {}, acc1 = {}, acc2 = {}, acc3 = {};
    for (int k0 = 0; k0 < 4608; k0 += 32) {
        int kb = k0 + ((lane >> 4) << 3);
        bf16x16 a;
#pragma unroll
        for (int g = 0; g < 2; ++g) {
            int kg = kb + g * 16;            // 8-elem group start, within one tap
            int tap = kg >> 9, ic = kg & 511;
            int ih = 2 * oh + tap / 3 - 1;
            int iw = 2 * ow + tap % 3 - 1;
            bf16x8 part = {};
            if ((unsigned)ih < 16u && (unsigned)iw < 16u) {
                const __bf16* p = Y + ((size_t)b * NT + ih * 16 + iw) * NED + ic;
                part = *reinterpret_cast<const bf16x8*>(p);
            }
#pragma unroll
            for (int i = 0; i < 8; ++i) a[g * 8 + i] = part[i];
        }
        bf16x16 b0 = load_bt_frag(WcT, 4608, n0,      k0, lane);
        bf16x16 b1 = load_bt_frag(WcT, 4608, n0 + 16, k0, lane);
        bf16x16 b2 = load_bt_frag(WcT, 4608, n0 + 32, k0, lane);
        bf16x16 b3 = load_bt_frag(WcT, 4608, n0 + 48, k0, lane);
        acc0 = wmma_bf16(a, b0, acc0);
        acc1 = wmma_bf16(a, b1, acc1);
        acc2 = wmma_bf16(a, b2, acc2);
        acc3 = wmma_bf16(a, b3, acc3);
    }
    store_tile(X, Xh, nullptr, NED, m0, n0,      lane, acc0);
    store_tile(X, Xh, nullptr, NED, m0, n0 + 16, lane, acc1);
    store_tile(X, Xh, nullptr, NED, m0, n0 + 32, lane, acc2);
    store_tile(X, Xh, nullptr, NED, m0, n0 + 48, lane, acc3);
}

// ---------------------------------------------------------------------------
// Distance GEMM + argmin: dist(m,n) = ||c_n||^2 - 2 x_m . c_n  (||x||^2 dropped).
// x tile staged into LDS via async DMA (ASYNCcnt) when available; each wave
// processes 64 codebook columns per outer step, reusing the LDS A fragment
// across 4 WMMAs. Running min packed as (orderedKey(dist)<<32)|n -> ds_min_u64.
// ---------------------------------------------------------------------------
__global__ void k_dist_argmin(const __bf16* __restrict__ Xh,
                              const __bf16* __restrict__ Cbh,
                              const float* __restrict__ cnorm,
                              int* __restrict__ minIdx, int* __restrict__ hist) {
    __shared__ __align__(16) __bf16 ldsX[16][NED];      // 16 KB
    __shared__ unsigned long long ldsMin[16];

    int m0 = blockIdx.x << 4;
    int t = threadIdx.x;
    if (t < 16) ldsMin[t] = ~0ull;

    // stage 16x512 bf16 tile (16 KB = 1024 x b128), 4 transfers per thread
#if defined(__gfx1250__) && __has_builtin(__builtin_amdgcn_global_load_async_to_lds_b128)
    {
        uintptr_t g = (uintptr_t)(Xh + (size_t)m0 * NED);
        unsigned  l = (unsigned)(uintptr_t)(&ldsX[0][0]);   // LDS offset = addr[31:0]
#pragma unroll
        for (int i = 0; i < 4; ++i) {
            __builtin_amdgcn_global_load_async_to_lds_b128(
                (__attribute__((address_space(1))) v4i_vs*)(g + (size_t)(t + i * 256) * 16),
                (__attribute__((address_space(3))) v4i_vs*)(uintptr_t)(l + (t + i * 256) * 16),
                0, 0);
        }
#if __has_builtin(__builtin_amdgcn_s_wait_asynccnt)
        __builtin_amdgcn_s_wait_asynccnt(0);
#endif
    }
#else
    {
        const uint4* src = reinterpret_cast<const uint4*>(Xh + (size_t)m0 * NED);
        uint4* dst = reinterpret_cast<uint4*>(&ldsX[0][0]);
#pragma unroll
        for (int i = 0; i < 4; ++i) dst[t + i * 256] = src[t + i * 256];
    }
#endif
    __syncthreads();

    int lane = t & 31, wave = t >> 5;
    unsigned long long best[8];
#pragma unroll
    for (int v = 0; v < 8; ++v) best[v] = ~0ull;

    for (int it = 0; it < 16; ++it) {
        int nBase = (it * 8 + wave) << 6;                // 64 columns per step
        v8f acc[4] = {};
#pragma unroll
        for (int k0 = 0; k0 < NED; k0 += 32) {
            int row = lane & 15;
            int kb  = k0 + ((lane >> 4) << 3);
            bf16x8 lo = *reinterpret_cast<const bf16x8*>(&ldsX[row][kb]);
            bf16x8 hi = *reinterpret_cast<const bf16x8*>(&ldsX[row][kb + 16]);
            bf16x16 a = frag_join(lo, hi);
#pragma unroll
            for (int nt = 0; nt < 4; ++nt) {
                bf16x16 b = load_bt_frag(Cbh, NED, nBase + nt * 16, k0, lane);
                acc[nt] = wmma_bf16(a, b, acc[nt]);
            }
        }
#pragma unroll
        for (int nt = 0; nt < 4; ++nt) {
            int col = nBase + nt * 16 + (lane & 15);
            float cn = cnorm[col];
#pragma unroll
            for (int v = 0; v < 8; ++v) {
                float dist = cn - 2.0f * acc[nt][v];
                unsigned u = __float_as_uint(dist);
                unsigned key = (u & 0x80000000u) ? ~u : (u | 0x80000000u);
                unsigned long long pk = ((unsigned long long)key << 32) | (unsigned)col;
                if (pk < best[v]) best[v] = pk;
            }
        }
    }

#pragma unroll
    for (int v = 0; v < 8; ++v) {
        int row = ((lane >> 4) << 3) + v;
        atomicMin(&ldsMin[row], best[v]);                // ds_min_u64
    }
    __syncthreads();

    if (t < 16) {
        int idx = (int)(unsigned)(ldsMin[t] & 0xFFFFFFFFu);
        minIdx[m0 + t] = idx;
        atomicAdd(&hist[idx], 1);
    }
}

// ---------------------------------------------------------------------------
// quantized = x + (||x - cb[idx]|| / ||rand|| + eps) * rand   (block per row)
// ---------------------------------------------------------------------------
__global__ void k_quantize(const float* __restrict__ X,
                           const float* __restrict__ Cb,
                           const float* __restrict__ rnd,
                           const int* __restrict__ minIdx,
                           float* __restrict__ Q) {
    __shared__ float sres[256], srnd[256];
    __shared__ float sscale;
    int r = blockIdx.x, t = threadIdx.x;
    int idx = minIdx[r];
    const float* xr = X   + (size_t)r * NED;
    const float* cr = Cb  + (size_t)idx * NED;
    const float* rr = rnd + (size_t)r * NED;
    float a = 0.f, b = 0.f;
    for (int e = t; e < NED; e += 256) {
        float d = xr[e] - cr[e]; a += d * d;
        float rv = rr[e];        b += rv * rv;
    }
    sres[t] = a; srnd[t] = b; __syncthreads();
    for (int s = 128; s > 0; s >>= 1) {
        if (t < s) { sres[t] += sres[t + s]; srnd[t] += srnd[t + s]; }
        __syncthreads();
    }
    if (t == 0) sscale = sqrtf(sres[0]) / sqrtf(srnd[0]) + 1e-12f;
    __syncthreads();
    for (int e = t; e < NED; e += 256) Q[(size_t)r * NED + e] = xr[e] + sscale * rr[e];
}

// raw-reshape gather for decode: A2[b*64+c][e] = Q_flat[b*32768 + e*64 + c]
__global__ void k_gather_a2(const float* __restrict__ Q, __bf16* __restrict__ A2) {
    int i = blockIdx.x * blockDim.x + threadIdx.x;
    int m = i >> 9, e = i & 511;
    int b = m >> 6, c = m & 63;
    A2[i] = (__bf16)Q[(size_t)b * 32768 + (size_t)e * 64 + c];
}

// perplexity + copy histogram / indices to output
__global__ void k_finalize(const int* __restrict__ hist, const int* __restrict__ minIdx,
                           float* __restrict__ outP, int* __restrict__ outHist,
                           int* __restrict__ outIdx) {
    __shared__ float ssum[256];
    int t = threadIdx.x;
    float s = 0.f;
    for (int i = t; i < NNE; i += 256) {
        int h = hist[i];
        outHist[i] = h;
        float p = (float)h * (1.0f / (float)NROWS);
        s += p * logf(p + 1e-12f);
    }
    ssum[t] = s; __syncthreads();
    for (int st = 128; st > 0; st >>= 1) {
        if (t < st) ssum[t] += ssum[t + st];
        __syncthreads();
    }
    if (t == 0) *outP = expf(-ssum[0]);
    for (int i = t; i < NROWS; i += 256) outIdx[i] = minIdx[i];
}

// ---------------------------------------------------------------------------
// Launch
// ---------------------------------------------------------------------------
static inline size_t align256(size_t x) { return (x + 255) & ~(size_t)255; }

extern "C" void kernel_launch(void* const* d_in, const int* in_sizes, int n_in,
                              void* d_out, int out_size, void* d_ws, size_t ws_size,
                              hipStream_t stream) {
    const float* in_first  = (const float*)d_in[0];
    const float* in_last   = (const float*)d_in[1];
    const float* W_in      = (const float*)d_in[2];
    // d_in[3] = b_in (cancels in last-first), d_in[5] = conv_b (cancels)
    const float* conv_w    = (const float*)d_in[4];
    const float* W_out     = (const float*)d_in[6];
    const float* b_out     = (const float*)d_in[7];
    const float* codebooks = (const float*)d_in[8];
    const float* rnd       = (const float*)d_in[9];

    char* ws = (char*)d_ws;
    size_t off = 0;
    auto carve = [&](size_t bytes) { void* p = ws + off; off += align256(bytes); return p; };
    __bf16* diffH = (__bf16*)carve((size_t)NB * NT * NDIM * 2);
    __bf16* WinT  = (__bf16*)carve((size_t)NED * NDIM * 2);
    __bf16* Y     = (__bf16*)carve((size_t)NB * NT * NED * 2);
    __bf16* WcT   = (__bf16*)carve((size_t)NED * 4608 * 2);
    float*  X     = (float*) carve((size_t)NROWS * NED * 4);
    __bf16* Xh    = (__bf16*)carve((size_t)NROWS * NED * 2);
    __bf16* Cbh   = (__bf16*)carve((size_t)NNE * NED * 2);
    float*  cnorm = (float*) carve((size_t)NNE * 4);
    int*    minIdx= (int*)   carve((size_t)NROWS * 4);
    int*    hist  = (int*)   carve((size_t)NNE * 4);
    float*  Q     = (float*) carve((size_t)NROWS * NED * 4);
    __bf16* A2    = (__bf16*)carve((size_t)NROWS * NED * 2);
    __bf16* WoutT = (__bf16*)carve((size_t)NDIM * NED * 2);
    (void)ws_size; (void)n_in; (void)in_sizes; (void)out_size;

    float* outF    = (float*)d_out;                 // [4096,1024] decode output
    float* outP    = outF + (size_t)NROWS * NDIM;   // perplexity scalar
    int*   outHist = (int*)(outP + 1);              // codebooks_used [8192]
    int*   outIdx  = outHist + NNE;                 // min_indices [4096]

    // prep / conversion (biases cancel in the encode difference)
    k_diff_cvt <<<65536, 256, 0, stream>>>(in_last, in_first, diffH, NB * NT * NDIM);
    k_cvt_winT <<<2048, 256, 0, stream>>>(W_in, WinT);
    k_cvt_wcT  <<<9216, 256, 0, stream>>>(conv_w, WcT);
    k_cvt_woutT<<<2048, 256, 0, stream>>>(W_out, WoutT);
    k_cvt_cb   <<<32, 256, 0, stream>>>(codebooks, Cbh, cnorm);
    k_zero_hist<<<32, 256, 0, stream>>>(hist);

    // GEMM1: diff @ W_in -> Y bf16 [16384,512]; waves = 1024 * 8
    k_gemm_bf16<<<1024, 256, 0, stream>>>(diffH, WinT, nullptr, Y, nullptr,
                                          NB * NT, NED, NDIM);

    // conv as implicit GEMM -> X fp32 / Xh bf16 [4096,512]; waves = 256 * 8
    k_conv_gemm<<<256, 256, 0, stream>>>(Y, WcT, X, Xh);

    // distances + argmin + histogram
    k_dist_argmin<<<NROWS / 16, 256, 0, stream>>>(Xh, Cbh, cnorm, minIdx, hist);

    // noise substitution quantization
    k_quantize<<<NROWS, 256, 0, stream>>>(X, codebooks, rnd, minIdx, Q);

    // decode: raw-reshape gather then GEMM2 into d_out; waves = 256 * 16
    k_gather_a2<<<8192, 256, 0, stream>>>(Q, A2);
    k_gemm_bf16<<<512, 256, 0, stream>>>(A2, WoutT, outF, nullptr, b_out,
                                         NROWS, NDIM, NED);

    k_finalize<<<1, 256, 0, stream>>>(hist, minIdx, outP, outHist, outIdx);
}
